// PointNetSetAbstraction_7705171329406
// MI455X (gfx1250) — compile-verified, matrix operand order checked by reading, not run on hardware
//
#include <hip/hip_runtime.h>
#include <hip/hip_bf16.h>

// ---------------- problem constants ----------------
#define BATCH   4
#define NPTS    8192
#define C_IN    64
#define MCENT   2048
#define KNN_K   32
#define NROWS   (BATCH * MCENT * KNN_K)   // 262144 GEMM rows
#define NCENT   (BATCH * MCENT)           // 8192 centroids
#define K0PAD   96                        // 67 padded to 3*32
#define EPS     1e-5f

typedef __attribute__((ext_vector_type(16))) _Float16 v16h;
typedef __attribute__((ext_vector_type(8)))  float    v8f;

// ============================================================
// 1) Farthest point sampling: one block per cloud.
//    Per-thread points cached in registers; dist array in LDS.
// ============================================================
__global__ __launch_bounds__(256) void fps_kernel(const float* __restrict__ p,
                                                  int* __restrict__ idx) {
    const int b = blockIdx.x;
    const int t = threadIdx.x;
    const float* pb = p + (size_t)b * NPTS * 3;

    float lx[32], ly[32], lz[32];
#pragma unroll
    for (int k = 0; k < 32; ++k) {
        int i = t + k * 256;
        lx[k] = pb[i * 3 + 0];
        ly[k] = pb[i * 3 + 1];
        lz[k] = pb[i * 3 + 2];
    }

    __shared__ float ds[NPTS];
    __shared__ float rv[256];
    __shared__ int   ri[256];
    __shared__ float cxs, cys, czs;

    const float x0 = pb[0], y0 = pb[1], z0 = pb[2];
#pragma unroll
    for (int k = 0; k < 32; ++k) {
        float dx = lx[k] - x0, dy = ly[k] - y0, dz = lz[k] - z0;
        ds[t + k * 256] = dx * dx + dy * dy + dz * dz;
    }
    if (t == 0) idx[b * MCENT] = 0;
    __syncthreads();

    for (int it = 1; it < MCENT; ++it) {
        // block argmax, first-occurrence tie-break (matches jnp.argmax)
        float bv = -1.0f; int bi = 0;
#pragma unroll
        for (int k = 0; k < 32; ++k) {
            int i = t + k * 256;
            float v = ds[i];
            if (v > bv) { bv = v; bi = i; }
        }
        rv[t] = bv; ri[t] = bi;
        __syncthreads();
        for (int st = 128; st > 0; st >>= 1) {
            if (t < st) {
                float ov = rv[t + st]; int oi = ri[t + st];
                if (ov > rv[t] || (ov == rv[t] && oi < ri[t])) { rv[t] = ov; ri[t] = oi; }
            }
            __syncthreads();
        }
        if (t == 0) {
            int j = ri[0];
            idx[b * MCENT + it] = j;
            cxs = pb[j * 3 + 0]; cys = pb[j * 3 + 1]; czs = pb[j * 3 + 2];
        }
        __syncthreads();
        const float qx = cxs, qy = cys, qz = czs;
#pragma unroll
        for (int k = 0; k < 32; ++k) {
            int i = t + k * 256;
            float dx = lx[k] - qx, dy = ly[k] - qy, dz = lz[k] - qz;
            float nd = dx * dx + dy * dy + dz * dz;
            float od = ds[i];
            ds[i] = nd < od ? nd : od;
        }
        __syncthreads();
    }
}

// ============================================================
// 2) Gather centroid coords; also emit n_p and n_o outputs.
// ============================================================
__global__ void gather_cents(const float* __restrict__ p, const int* __restrict__ idx,
                             float* __restrict__ cents, float* __restrict__ out_p,
                             float* __restrict__ out_o) {
    int i = blockIdx.x * 256 + threadIdx.x;
    if (i >= NCENT) return;
    int b = i / MCENT;
    int j = idx[i];
    const float* src = p + ((size_t)b * NPTS + j) * 3;
#pragma unroll
    for (int c = 0; c < 3; ++c) {
        float v = src[c];
        cents[i * 3 + c] = v;
        out_p[i * 3 + c]  = v;
    }
    if (i < BATCH) out_o[i] = (float)((i + 1) * MCENT);
}

// ============================================================
// 3) KNN (exact 32-NN via iterative argmin extraction) + group:
//    emits fp16 rows [rel_xyz | feats | pad] of width K0PAD.
// ============================================================
__global__ __launch_bounds__(256) void knn_group(const float* __restrict__ p,
                                                 const float* __restrict__ x,
                                                 const float* __restrict__ cents,
                                                 _Float16* __restrict__ A0) {
    const int cent = blockIdx.x;
    const int t = threadIdx.x;
    const int b = cent / MCENT;
    const float* pb = p + (size_t)b * NPTS * 3;

    __shared__ float ds[NPTS];
    __shared__ float rv[256];
    __shared__ int   ri[256];
    __shared__ int   nn[KNN_K];

    const float qx = cents[cent * 3 + 0];
    const float qy = cents[cent * 3 + 1];
    const float qz = cents[cent * 3 + 2];

    for (int i = t; i < NPTS; i += 256) {
        float dx = pb[i * 3 + 0] - qx;
        float dy = pb[i * 3 + 1] - qy;
        float dz = pb[i * 3 + 2] - qz;
        ds[i] = dx * dx + dy * dy + dz * dz;
    }
    __syncthreads();

    for (int kk = 0; kk < KNN_K; ++kk) {
        float bv = 3.4e38f; int bi = 0;
        for (int i = t; i < NPTS; i += 256) {
            float v = ds[i];
            if (v < bv) { bv = v; bi = i; }
        }
        rv[t] = bv; ri[t] = bi;
        __syncthreads();
        for (int st = 128; st > 0; st >>= 1) {
            if (t < st) {
                float ov = rv[t + st]; int oi = ri[t + st];
                if (ov < rv[t] || (ov == rv[t] && oi < ri[t])) { rv[t] = ov; ri[t] = oi; }
            }
            __syncthreads();
        }
        if (t == 0) { nn[kk] = ri[0]; ds[ri[0]] = 3.4e38f; }
        __syncthreads();
    }

    const float* xb = x + (size_t)b * NPTS * C_IN;
    for (int e = t; e < KNN_K * K0PAD; e += 256) {
        int k = e / K0PAD, c = e % K0PAD;
        int j = nn[k];
        float v;
        if (c < 3)        v = pb[j * 3 + c] - cents[cent * 3 + c];
        else if (c < 67)  v = xb[(size_t)j * C_IN + (c - 3)];
        else              v = 0.0f;
        A0[(size_t)(cent * KNN_K + k) * K0PAD + c] = (_Float16)v;
    }
}

// ============================================================
// 4) Weight prep: transpose+pad W [Kreal x Cout] f32 -> Wt [Cout x Kpad] fp16
// ============================================================
__global__ void prep_w(const float* __restrict__ W, _Float16* __restrict__ Wt,
                       int Kreal, int Kpad, int Cout) {
    int i = blockIdx.x * 256 + threadIdx.x;
    if (i >= Kpad * Cout) return;
    int co = i / Kpad, k = i % Kpad;
    Wt[i] = (k < Kreal) ? (_Float16)W[(size_t)k * Cout + co] : (_Float16)0.0f;
}

// ============================================================
// 5) WMMA GEMM: Y[NROWS x (16*NT)] = A[NROWS x (32*KS)] * Wt^T
//    - All B (weight) fragments preloaded into registers once per
//      wave and reused across RT row tiles -> back-to-back WMMA
//      chains with a single A-load clause + wait per row tile.
//    A frag: lane(0-15)=row, lane/16 selects K-half -> 32B contiguous load.
//    B frag: Wt stored [Cout][Kpad] -> same contiguous pattern per lane/column.
// ============================================================
template <int KS, int NT, int RT>
__global__ __launch_bounds__(256) void gemm_wmma(const _Float16* __restrict__ A,
                                                 const _Float16* __restrict__ Bt,
                                                 _Float16* __restrict__ Y) {
    const int lane  = threadIdx.x & 31;
    const int wave  = threadIdx.x >> 5;
    const int tile0 = (blockIdx.x * 8 + wave) * RT;
    const int KP = KS * 32;
    const int CO = NT * 16;
    const int l15  = lane & 15;
    const int koff = (lane >> 4) * 16;

    // ---- preload all weight fragments into registers (hoisted clause) ----
    v16h bf[KS * NT];
#pragma unroll
    for (int ks = 0; ks < KS; ++ks) {
#pragma unroll
        for (int n = 0; n < NT; ++n) {
            bf[ks * NT + n] =
                *(const v16h*)(Bt + (size_t)(n * 16 + l15) * KP + ks * 32 + koff);
        }
    }

    for (int rt = 0; rt < RT; ++rt) {
        const int tile = tile0 + rt;
        const int rowA = tile * 16 + l15;

        // one clause of contiguous A loads for this row tile
        v16h a[KS];
#pragma unroll
        for (int ks = 0; ks < KS; ++ks) {
            a[ks] = *(const v16h*)(A + (size_t)rowA * KP + ks * 32 + koff);
        }

        v8f acc[NT];
#pragma unroll
        for (int n = 0; n < NT; ++n) { v8f z = {}; acc[n] = z; }

        // dense WMMA chain, no memory waits in between
#pragma unroll
        for (int ks = 0; ks < KS; ++ks) {
#pragma unroll
            for (int n = 0; n < NT; ++n) {
                acc[n] = __builtin_amdgcn_wmma_f32_16x16x32_f16(
                    false, a[ks], false, bf[ks * NT + n], (short)0, acc[n], false, false);
            }
        }

        const int rbase = tile * 16 + (lane >> 4) * 8;
#pragma unroll
        for (int n = 0; n < NT; ++n) {
#pragma unroll
            for (int r = 0; r < 8; ++r) {
                Y[(size_t)(rbase + r) * CO + n * 16 + l15] = (_Float16)acc[n][r];
            }
        }
    }
}

// ============================================================
// 6) BatchNorm stats (deterministic two-stage) + fused BN+ReLU
// ============================================================
template <int C>
__global__ __launch_bounds__(256) void bn_stats1(const _Float16* __restrict__ Y,
                                                 float* __restrict__ part, int rowsPerBlk) {
    constexpr int RPP = 256 / C;
    const int c  = threadIdx.x % C;
    const int rl = threadIdx.x / C;
    const int r0 = blockIdx.x * rowsPerBlk;
    float s = 0.0f, sq = 0.0f;
    for (int r = r0 + rl; r < r0 + rowsPerBlk; r += RPP) {
        float v = (float)Y[(size_t)r * C + c];
        s += v; sq += v * v;
    }
    __shared__ float ls[256], lq[256];
    ls[threadIdx.x] = s; lq[threadIdx.x] = sq;
    __syncthreads();
    for (int st = RPP / 2; st > 0; st >>= 1) {
        if (rl < st) {
            ls[threadIdx.x] += ls[threadIdx.x + st * C];
            lq[threadIdx.x] += lq[threadIdx.x + st * C];
        }
        __syncthreads();
    }
    if (rl == 0) {
        part[(size_t)blockIdx.x * 2 * C + c]     = ls[c];
        part[(size_t)blockIdx.x * 2 * C + C + c] = lq[c];
    }
}

template <int C>
__global__ void bn_stats2(const float* __restrict__ part, const float* __restrict__ g,
                          const float* __restrict__ be, float* __restrict__ scsh,
                          int nblk, float inv_n) {
    int c = threadIdx.x;
    if (c >= C) return;
    float s = 0.0f, sq = 0.0f;
    for (int bk = 0; bk < nblk; ++bk) {
        s  += part[(size_t)bk * 2 * C + c];
        sq += part[(size_t)bk * 2 * C + C + c];
    }
    float mean = s * inv_n;
    float var  = sq * inv_n - mean * mean;
    float inv  = rsqrtf(var + EPS);
    float sc   = g[c] * inv;
    scsh[c]     = sc;
    scsh[C + c] = be[c] - mean * sc;
}

template <int C>
__global__ __launch_bounds__(256) void bn_relu(const _Float16* __restrict__ Y,
                                               const float* __restrict__ scsh,
                                               _Float16* __restrict__ Aout) {
    size_t i = (size_t)blockIdx.x * 256 + threadIdx.x;
    int c = (int)(i % C);
    float v = (float)Y[i] * scsh[c] + scsh[C + c];
    Aout[i] = (_Float16)fmaxf(v, 0.0f);
}

// ============================================================
// 7) Max-pool over K neighbors -> n_x
// ============================================================
__global__ void maxpool_k(const _Float16* __restrict__ A3, float* __restrict__ out_x) {
    int cent = blockIdx.x;
    int c = threadIdx.x;  // 128
    float m = -3.4e38f;
#pragma unroll
    for (int k = 0; k < KNN_K; ++k) {
        float v = (float)A3[(size_t)(cent * KNN_K + k) * 128 + c];
        m = fmaxf(m, v);
    }
    out_x[(size_t)cent * 128 + c] = m;
}

// ============================================================
// Launcher
// ============================================================
extern "C" void kernel_launch(void* const* d_in, const int* in_sizes, int n_in,
                              void* d_out, int out_size, void* d_ws, size_t ws_size,
                              hipStream_t stream) {
    const float* p  = (const float*)d_in[0];
    const float* x  = (const float*)d_in[1];
    // d_in[2] = o (int32) — equal-size clouds, constants hardcoded
    const float* W0 = (const float*)d_in[3];
    const float* g0 = (const float*)d_in[4];
    const float* be0= (const float*)d_in[5];
    const float* W1 = (const float*)d_in[6];
    const float* g1 = (const float*)d_in[7];
    const float* be1= (const float*)d_in[8];
    const float* W2 = (const float*)d_in[9];
    const float* g2 = (const float*)d_in[10];
    const float* be2= (const float*)d_in[11];

    float* out_p = (float*)d_out;                       // 8192*3
    float* out_x = out_p + (size_t)NCENT * 3;           // 8192*128
    float* out_o = out_x + (size_t)NCENT * 128;         // 4

    // workspace carve-up (256B aligned)
    char* ws = (char*)d_ws;
    size_t off = 0;
    auto carve = [&](size_t bytes) { char* r = ws + off; off = (off + bytes + 255) & ~(size_t)255; return r; };
    int*      idx   = (int*)     carve((size_t)NCENT * 4);
    float*    cents = (float*)   carve((size_t)NCENT * 3 * 4);
    _Float16* Wt0   = (_Float16*)carve((size_t)64 * K0PAD * 2);
    _Float16* Wt1   = (_Float16*)carve((size_t)64 * 64 * 2);
    _Float16* Wt2   = (_Float16*)carve((size_t)128 * 64 * 2);
    float*    part  = (float*)   carve((size_t)512 * 2 * 128 * 4);
    float*    scsh  = (float*)   carve((size_t)2 * 128 * 4);
    _Float16* ACT   = (_Float16*)carve((size_t)NROWS * 128 * 2);   // activations (ping)
    _Float16* YB    = (_Float16*)carve((size_t)NROWS * 128 * 2);   // gemm outputs (pong)
    (void)ws_size; (void)in_sizes; (void)n_in; (void)out_size;

    // ---- sampling + grouping ----
    fps_kernel<<<BATCH, 256, 0, stream>>>(p, idx);
    gather_cents<<<(NCENT + 255) / 256, 256, 0, stream>>>(p, idx, cents, out_p, out_o);
    knn_group<<<NCENT, 256, 0, stream>>>(p, x, cents, ACT);

    // ---- weight prep ----
    prep_w<<<(K0PAD * 64 + 255) / 256, 256, 0, stream>>>(W0, Wt0, 67, K0PAD, 64);
    prep_w<<<(64 * 64 + 255) / 256, 256, 0, stream>>>(W1, Wt1, 64, 64, 64);
    prep_w<<<(64 * 128 + 255) / 256, 256, 0, stream>>>(W2, Wt2, 64, 64, 128);

    // 16384 row tiles, 8 waves/block, RT=4 tiles/wave -> 512 blocks
    constexpr int RT = 4;
    const int gemmGrid = NROWS / 16 / 8 / RT;
    const float invN = 1.0f / (float)NROWS;

    // ---- layer 0: (NROWS x 96) @ (96 x 64) ----
    gemm_wmma<3, 4, RT><<<gemmGrid, 256, 0, stream>>>(ACT, Wt0, YB);
    bn_stats1<64><<<512, 256, 0, stream>>>(YB, part, NROWS / 512);
    bn_stats2<64><<<1, 64, 0, stream>>>(part, g0, be0, scsh, 512, invN);
    bn_relu<64><<<(int)((size_t)NROWS * 64 / 256), 256, 0, stream>>>(YB, scsh, ACT);

    // ---- layer 1: (NROWS x 64) @ (64 x 64) ----
    gemm_wmma<2, 4, RT><<<gemmGrid, 256, 0, stream>>>(ACT, Wt1, YB);
    bn_stats1<64><<<512, 256, 0, stream>>>(YB, part, NROWS / 512);
    bn_stats2<64><<<1, 64, 0, stream>>>(part, g1, be1, scsh, 512, invN);
    bn_relu<64><<<(int)((size_t)NROWS * 64 / 256), 256, 0, stream>>>(YB, scsh, ACT);

    // ---- layer 2: (NROWS x 64) @ (64 x 128) ----
    gemm_wmma<2, 8, RT><<<gemmGrid, 256, 0, stream>>>(ACT, Wt2, YB);
    bn_stats1<128><<<512, 256, 0, stream>>>(YB, part, NROWS / 512);
    bn_stats2<128><<<1, 128, 0, stream>>>(part, g2, be2, scsh, 512, invN);
    bn_relu<128><<<(int)((size_t)NROWS * 128 / 256), 256, 0, stream>>>(YB, scsh, ACT);

    // ---- pooling ----
    maxpool_k<<<NCENT, 128, 0, stream>>>(ACT, out_x);
}
